// DEDICOMDecoder_62612033241832
// MI455X (gfx1250) — matrix operand order repaired and checked by reading, not compiled, not run on hardware
//
#include <hip/hip_runtime.h>
#include <hip/hip_bf16.h>

typedef __attribute__((ext_vector_type(16))) _Float16 v16h;
typedef __attribute__((ext_vector_type(8)))  float    v8f;

#define N_ROWS  500000
#define DDIM    128
#define KREL    8
#define RTILE   4                 // 4 row-tiles of 16 rows -> 64 rows per wave-iteration
#define HSTR    136               // padded column stride (f16 elems): 4-bank step, 16B aligned
#define NCOLS   (KREL * DDIM)     // 1024 combined columns (k,a)
#define THREADS 512
#define WAVES   (THREADS / 32)
#define BLOCKS  256

// ds_swizzle group-of-32 XOR exchange: offset = xor_mask<<10 | or_mask<<5 | and_mask
#define SWZ_XOR(x, m) \
    __int_as_float(__builtin_amdgcn_ds_swizzle(__float_as_int(x), (((m) << 10) | 0x1F)))

// out[k*N + i] = sigmoid( row_i^T * M_k * col_i ),  M_k[a,b] = d_k[a]*G[a,b]*d_k[b]
// Computed as W = Vtile(16x128,f16) @ B(128x1024,f16 in LDS), epilogue row-dot with Utile.
__global__ __launch_bounds__(THREADS, 1)
void dedicom_wmma_kernel(const float* __restrict__ U,    // inputs_row [N,D]
                         const float* __restrict__ V,    // inputs_col [N,D]
                         const float* __restrict__ G,    // [D,D]
                         const float* __restrict__ dv,   // [K,D]
                         float* __restrict__ out)        // [K,N]
{
    // B-matrix for the combined GEMM, column-major: H[col*HSTR + b] = M_k[a,b],
    // col = k*128 + a.  1024*136*2 = 272 KB of the WGP's 320 KB LDS.
    __shared__ _Float16 H[NCOLS * HSTR];

    const int tid = threadIdx.x;

    // ---- build all 8 M_k into LDS (G is L2-resident, read coalesced over b) ----
    for (int idx = tid; idx < NCOLS * DDIM; idx += THREADS) {
        const int b   = idx & (DDIM - 1);
        const int col = idx >> 7;
        const int a   = col & (DDIM - 1);
        const int k   = col >> 7;
        const float val = dv[k * DDIM + a] * G[a * DDIM + b] * dv[k * DDIM + b];
        H[col * HSTR + b] = (_Float16)val;
    }
    __syncthreads();

    const int lane = tid & 31;
    const int wave = tid >> 5;
    const int half = lane >> 4;      // ISA A/B lane-half
    const int l16  = lane & 15;
    const bool up4 = (l16 & 4) != 0;
    const bool up2 = (l16 & 2) != 0;
    const bool up1 = (l16 & 1) != 0;
    const int hlane = l16 * HSTR + half * 8;    // per-lane B-fragment base in H

    const int nSets  = (N_ROWS + 16 * RTILE - 1) / (16 * RTILE);   // 7813
    const int gwave  = blockIdx.x * WAVES + wave;
    const int nWaves = gridDim.x * WAVES;

    for (int set = gwave; set < nSets; set += nWaves) {
        const int row0 = set * (16 * RTILE);

        // 32-bit element offsets (GVS addressing: SGPR base + u32 voffset + imm).
        // Tail clamp: N is a multiple of 32, not 64.
        unsigned vbase[RTILE];   // V row base for A loads (row = rbase + l16)
        unsigned ubase[RTILE];   // U row base for epilogue (row = rbase + half*8 + m)
        #pragma unroll
        for (int r = 0; r < RTILE; ++r) {
            int rb = row0 + r * 16;
            if (rb + 16 > N_ROWS) rb = N_ROWS - 16;
            vbase[r] = (unsigned)(rb + l16) * DDIM;
            ubase[r] = (unsigned)(rb + half * 8) * DDIM + (unsigned)l16;
        }

        // ---- A fragments: col-input tile, f32 -> f16, ISA 16x32 A layout ----
        // lane m (both halves) holds row m; half 0: K = kb*32 + {0..7, 16..23},
        // half 1: K = kb*32 + {8..15, 24..31}.  Two b128 loads per chunk.
        v16h A[RTILE][4];
        #pragma unroll
        for (int r = 0; r < RTILE; ++r) {
            #pragma unroll
            for (int kb = 0; kb < 4; ++kb) {
                const float* p = V + (vbase[r] + kb * 32 + half * 8);
                const float4 f0 = *(const float4*)(p);
                const float4 f1 = *(const float4*)(p + 4);
                const float4 f2 = *(const float4*)(p + 16);
                const float4 f3 = *(const float4*)(p + 20);
                A[r][kb][0]  = (_Float16)f0.x;  A[r][kb][1]  = (_Float16)f0.y;
                A[r][kb][2]  = (_Float16)f0.z;  A[r][kb][3]  = (_Float16)f0.w;
                A[r][kb][4]  = (_Float16)f1.x;  A[r][kb][5]  = (_Float16)f1.y;
                A[r][kb][6]  = (_Float16)f1.z;  A[r][kb][7]  = (_Float16)f1.w;
                A[r][kb][8]  = (_Float16)f2.x;  A[r][kb][9]  = (_Float16)f2.y;
                A[r][kb][10] = (_Float16)f2.z;  A[r][kb][11] = (_Float16)f2.w;
                A[r][kb][12] = (_Float16)f3.x;  A[r][kb][13] = (_Float16)f3.y;
                A[r][kb][14] = (_Float16)f3.z;  A[r][kb][15] = (_Float16)f3.w;
            }
        }

        float Tacc[RTILE][KREL];
        #pragma unroll
        for (int r = 0; r < RTILE; ++r)
            #pragma unroll
            for (int k = 0; k < KREL; ++k) Tacc[r][k] = 0.0f;

        #pragma unroll 1
        for (int nb = 0; nb < 8; ++nb) {
            // U values for this 16-column group: column a = nb*16 + l16
            float Uv[RTILE][8];
            #pragma unroll
            for (int r = 0; r < RTILE; ++r)
                #pragma unroll
                for (int m = 0; m < 8; ++m)
                    Uv[r][m] = U[ubase[r] + (unsigned)(m * DDIM + nb * 16)];

            #pragma unroll
            for (int k = 0; k < KREL; ++k) {
                const int colblk = k * 8 + nb;          // global 16-col block
                const _Float16* bp = H + (hlane + colblk * (16 * HSTR));

                v8f acc[RTILE];
                #pragma unroll
                for (int r = 0; r < RTILE; ++r) {
                    v8f z = {0.f, 0.f, 0.f, 0.f, 0.f, 0.f, 0.f, 0.f};
                    acc[r] = z;
                }

                // double-buffered B fragments: load kb+1 while kb's WMMAs run
                union Bu { v16h v; int4 q[2]; };
                Bu Bb[2];
                Bb[0].q[0] = *(const int4*)(bp);
                Bb[0].q[1] = *(const int4*)(bp + 16);
                #pragma unroll
                for (int kb = 0; kb < 4; ++kb) {
                    if (kb < 3) {
                        const _Float16* bpn = bp + (kb + 1) * 32;
                        Bb[(kb + 1) & 1].q[0] = *(const int4*)(bpn);
                        Bb[(kb + 1) & 1].q[1] = *(const int4*)(bpn + 16);
                    }
                    #pragma unroll
                    for (int r = 0; r < RTILE; ++r)
                        acc[r] = __builtin_amdgcn_wmma_f32_16x16x32_f16(
                                     false, A[r][kb], false, Bb[kb & 1].v,
                                     (short)0, acc[r], false, false);
                }

                // epilogue: s[m'] = sum_{n'=0..15} U[row,nb*16+n'] * W[row, col n'].
                // Distributing butterfly (bits 2,1,0) then one xor-8 fold: lane n'
                // ends holding the sum for row (8*half + (n'&7)).  8 ds_swizzle total.
                #pragma unroll
                for (int r = 0; r < RTILE; ++r) {
                    float p0 = acc[r][0] * Uv[r][0];
                    float p1 = acc[r][1] * Uv[r][1];
                    float p2 = acc[r][2] * Uv[r][2];
                    float p3 = acc[r][3] * Uv[r][3];
                    float p4 = acc[r][4] * Uv[r][4];
                    float p5 = acc[r][5] * Uv[r][5];
                    float p6 = acc[r][6] * Uv[r][6];
                    float p7 = acc[r][7] * Uv[r][7];

                    float k0 = up4 ? p4 : p0;   float s0 = up4 ? p0 : p4;
                    float k1 = up4 ? p5 : p1;   float s1 = up4 ? p1 : p5;
                    float k2 = up4 ? p6 : p2;   float s2 = up4 ? p2 : p6;
                    float k3 = up4 ? p7 : p3;   float s3 = up4 ? p3 : p7;
                    k0 += SWZ_XOR(s0, 4);
                    k1 += SWZ_XOR(s1, 4);
                    k2 += SWZ_XOR(s2, 4);
                    k3 += SWZ_XOR(s3, 4);

                    float j0 = up2 ? k2 : k0;   float t0 = up2 ? k0 : k2;
                    float j1 = up2 ? k3 : k1;   float t1 = up2 ? k1 : k3;
                    j0 += SWZ_XOR(t0, 2);
                    j1 += SWZ_XOR(t1, 2);

                    float w0 = up1 ? j1 : j0;   float u0 = up1 ? j0 : j1;
                    w0 += SWZ_XOR(u0, 1);

                    w0 += SWZ_XOR(w0, 8);       // fold the two n'-octets

                    Tacc[r][k] += w0;
                }
            }
        }

        // ---- sigmoid + store: lane (half, n'<8) owns row row0 + r*16 + 8*half + n' ----
        if ((lane & 8) == 0) {
            #pragma unroll
            for (int r = 0; r < RTILE; ++r) {
                const int rowr = row0 + r * 16;
                if (rowr + 16 <= N_ROWS) {
                    const int mrow = rowr + half * 8 + (l16 & 7);
                    #pragma unroll
                    for (int k = 0; k < KREL; ++k) {
                        const float s = Tacc[r][k];
                        out[(size_t)k * N_ROWS + mrow] = 1.0f / (1.0f + __expf(-s));
                    }
                }
            }
        }
    }
}

extern "C" void kernel_launch(void* const* d_in, const int* in_sizes, int n_in,
                              void* d_out, int out_size, void* d_ws, size_t ws_size,
                              hipStream_t stream) {
    (void)in_sizes; (void)n_in; (void)out_size; (void)d_ws; (void)ws_size;
    const float* U  = (const float*)d_in[0];   // inputs_row
    const float* V  = (const float*)d_in[1];   // inputs_col
    const float* G  = (const float*)d_in[2];   // global_interaction
    const float* dv = (const float*)d_in[3];   // local_variation
    float* out = (float*)d_out;

    dedicom_wmma_kernel<<<dim3(BLOCKS), dim3(THREADS), 0, stream>>>(U, V, G, dv, out);
}
